// CascadeRefinementVolume_65171833749963
// MI455X (gfx1250) — matrix operand order repaired and checked by reading, not compiled, not run on hardware
//
#include <hip/hip_runtime.h>
#include <hip/hip_bf16.h>
#include <math.h>

// ---------------- problem constants ----------------
#define BB   4
#define CC   64
#define HH   160
#define WW   320
#define GG   8
#define HR   4
#define AG   24
#define DDZ  9                    // D = 2*HR+1
#define HWXY (HH*WW)              // 51200
#define DHW  (DDZ*HWXY)           // 460800
#define KPAD_MAX 672              // 24*27 rounded up to 32

// halo-padded activation layout: [c][dz+1][y+1][x+1], pads are zero
#define PW   322
#define PH   162
#define PD   11
#define PHW  (PW*PH)              // 52164
#define PDHW (PD*PHW)             // 573804

typedef __bf16 bf16;
typedef __attribute__((ext_vector_type(16))) __bf16 v16bf;
typedef __attribute__((ext_vector_type(8)))  float  v8f;

// =====================================================================
// 0) zero the padded activation arena (pads must read as 0)
// =====================================================================
__global__ __launch_bounds__(256) void zero_arena_kernel(uint4* __restrict__ p,
                                                         long n16)
{
  long i = (long)blockIdx.x * blockDim.x + threadIdx.x;
  if (i < n16) p[i] = make_uint4(0u, 0u, 0u, 0u);
}

// =====================================================================
// 1) warp-correlation cost volume -> padded (B,G,D,H,W) bf16
// =====================================================================
__global__ __launch_bounds__(256) void warp_corr_kernel(
    const float* __restrict__ fL, const float* __restrict__ fR,
    const float* __restrict__ dC, bf16* __restrict__ cvp)
{
  int idx = blockIdx.x * blockDim.x + threadIdx.x;          // over B*D*H*W
  if (idx >= BB * DDZ * HWXY) return;
  int w = idx % WW;  int t = idx / WW;
  int h = t % HH;    t /= HH;
  int dd = t % DDZ;  int b = t / DDZ;

  float dtot = dC[(size_t)b * HWXY + h * WW + w] + (float)(dd - HR);
  float xs   = fminf(fmaxf((float)w - dtot, 0.f), (float)(WW - 1));
  int   x0   = (int)floorf(xs);
  int   x1   = min(x0 + 1, WW - 1);
  float wt   = xs - (float)x0;

  float acc[GG];
#pragma unroll
  for (int g = 0; g < GG; ++g) acc[g] = 0.f;

  const size_t rowb = (size_t)b * CC * HWXY + (size_t)h * WW;
  for (int c = 0; c < CC; ++c) {
    size_t row = rowb + (size_t)c * HWXY;
    float fl = fL[row + w];
    float fr = fR[row + x0] * (1.f - wt) + fR[row + x1] * wt;
    acc[c >> 3] += fl * fr;
  }
  const size_t pb = (size_t)(dd + 1) * PHW + (size_t)(h + 1) * PW + (w + 1);
#pragma unroll
  for (int g = 0; g < GG; ++g)
    cvp[(size_t)(b * GG + g) * PDHW + pb] = (bf16)(acc[g] * 0.125f);
}

// =====================================================================
// 2) implicit-GEMM 3x3x3 conv via v_wmma_f32_16x16x32_bf16
//    wave tile: M=32 w-positions (two 16-row tiles, B reuse) x N=24
//    (two 16-col tiles).  K = CIN*27 in 32-chunks.
//    Halo-padded input => unconditional gathers, zero bounds checks.
//    K-padding rides on zeroed weight rows.  Fused GN partial sums.
// =====================================================================
__global__ __launch_bounds__(128) void conv3d_wmma_kernel(
    const bf16* __restrict__ Xp, const float* __restrict__ Wt,
    float* __restrict__ Y,
    float* __restrict__ stat_sum, float* __restrict__ stat_sq, int CIN)
{
  const int Kreal = CIN * 27;
  const int Kpad  = (Kreal + 31) & ~31;

  __shared__ bf16  lds_w[24 * KPAD_MAX];   // weights [n][Kpad], zero-padded K
  __shared__ int   lds_delta[KPAD_MAX];    // padded-layout addr delta per K
  __shared__ float ls[GG], ls2[GG];

  // ---- stage weights f32 -> bf16, contiguous-K rows (pad rows = 0)
  for (int n = 0; n < 24; ++n)
    for (int k = threadIdx.x; k < Kpad; k += blockDim.x)
      lds_w[n * Kpad + k] = (k < Kreal) ? (bf16)Wt[(size_t)n * Kreal + k]
                                        : (bf16)0.f;
  // ---- stage K -> address-delta table (divisions once per block)
  for (int k = threadIdx.x; k < Kpad; k += blockDim.x) {
    int delta = 0;
    if (k < Kreal) {
      int c  = k / 27;
      int rr = k - c * 27;
      int kd = rr / 9;
      int r9 = rr - kd * 9;
      int ky = r9 / 3;
      int kx = r9 - ky * 3;
      delta = c * PDHW + (kd - 1) * PHW + (ky - 1) * PW + (kx - 1);
    }
    lds_delta[k] = delta;
  }
  if (threadIdx.x < GG) { ls[threadIdx.x] = 0.f; ls2[threadIdx.x] = 0.f; }
  __syncthreads();

  const int lane = threadIdx.x & 31;
  const int wave = threadIdx.x >> 5;
  // 14400 tiles per batch (9*160*10), 4 waves/block -> 3600 blocks per batch
  const int blocks_per_b = 3600;
  const int b    = blockIdx.x / blocks_per_b;
  const int tile = (blockIdx.x % blocks_per_b) * 4 + wave;
  const int d    = tile / (HH * (WW / 32));
  const int r2   = tile % (HH * (WW / 32));
  const int h    = r2 / (WW / 32);
  const int w0   = (r2 % (WW / 32)) * 32;

  const int m    = lane & 15;              // A row / B column within tile
  const int half = lane >> 4;

  // per-lane base pointer into the padded tensor (center tap)
  const bf16* pbase = Xp + (size_t)b * CIN * PDHW + (size_t)(d + 1) * PHW +
                      (size_t)(h + 1) * PW + (w0 + m + 1);

  // B rows: clamp tile-1 row for lanes m>=8 (their columns are never stored)
  const bf16* wrow0 = &lds_w[m * Kpad];
  const bf16* wrow1 = &lds_w[min(16 + m, 23) * Kpad];

  v8f acc00 = {}, acc01 = {}, acc10 = {}, acc11 = {};

  for (int kc = 0; kc < Kpad; kc += 32) {
    const int kb0 = kc + 16 * half;        // K ids for j=0..7
    const int kb1 = kc + 16 * half + 8;    // K ids for j=8..15
    int4 dA = *(const int4*)&lds_delta[kb0];
    int4 dB = *(const int4*)&lds_delta[kb0 + 4];
    int4 dE = *(const int4*)&lds_delta[kb1];
    int4 dF = *(const int4*)&lds_delta[kb1 + 4];

    v16bf a0, a1;
#define DOJ(j, dv)                          \
    {                                       \
      a0[j] = pbase[(dv)];                  \
      a1[j] = pbase[(dv) + 16];             \
    }
    DOJ(0,  dA.x) DOJ(1,  dA.y) DOJ(2,  dA.z) DOJ(3,  dA.w)
    DOJ(4,  dB.x) DOJ(5,  dB.y) DOJ(6,  dB.z) DOJ(7,  dB.w)
    DOJ(8,  dE.x) DOJ(9,  dE.y) DOJ(10, dE.z) DOJ(11, dE.w)
    DOJ(12, dF.x) DOJ(13, dF.y) DOJ(14, dF.z) DOJ(15, dF.w)
#undef DOJ

    v16bf b0 = *(const v16bf*)&wrow0[kc + 16 * half];
    v16bf b1 = *(const v16bf*)&wrow1[kc + 16 * half];

    acc00 = __builtin_amdgcn_wmma_f32_16x16x32_bf16(false, a0, false, b0,
                                                    (short)0, acc00, false, false);
    acc01 = __builtin_amdgcn_wmma_f32_16x16x32_bf16(false, a0, false, b1,
                                                    (short)0, acc01, false, false);
    acc10 = __builtin_amdgcn_wmma_f32_16x16x32_bf16(false, a1, false, b0,
                                                    (short)0, acc10, false, false);
    acc11 = __builtin_amdgcn_wmma_f32_16x16x32_bf16(false, a1, false, b1,
                                                    (short)0, acc11, false, false);
  }

  // ---- epilogue: store dense f32 conv output + fused GN partial sums
  float s0 = 0.f, q0 = 0.f, s1 = 0.f, q1 = 0.f;
  const size_t orow0 = ((size_t)(b * AG + m) * DDZ + d) * HWXY +
                       (size_t)h * WW + w0;
  const size_t orow1 = ((size_t)(b * AG + 16 + m) * DDZ + d) * HWXY +
                       (size_t)h * WW + w0;
#pragma unroll
  for (int i = 0; i < 8; ++i) {
    int mm = i + 8 * half;
    float v;
    v = acc00[i]; Y[orow0 + mm]      = v; s0 += v; q0 += v * v;
    v = acc10[i]; Y[orow0 + mm + 16] = v; s0 += v; q0 += v * v;
    if (m < 8) {
      v = acc01[i]; Y[orow1 + mm]      = v; s1 += v; q1 += v * v;
      v = acc11[i]; Y[orow1 + mm + 16] = v; s1 += v; q1 += v * v;
    }
  }
  atomicAdd(&ls[m / 3], s0);  atomicAdd(&ls2[m / 3], q0);
  if (m < 8) { atomicAdd(&ls[(16 + m) / 3], s1); atomicAdd(&ls2[(16 + m) / 3], q1); }
  __syncthreads();
  if (threadIdx.x < GG) {
    atomicAdd(&stat_sum[b * GG + threadIdx.x], ls[threadIdx.x]);
    atomicAdd(&stat_sq [b * GG + threadIdx.x], ls2[threadIdx.x]);
  }
}

// =====================================================================
// 3) GN stats helpers
// =====================================================================
__global__ void zero_stats_kernel(float* s) {
  if (threadIdx.x < 64) s[threadIdx.x] = 0.f;   // sum[32] + sumsq[32]
}

__global__ void finalize_stats_kernel(const float* __restrict__ sum,
                                      const float* __restrict__ sq,
                                      float* __restrict__ mean,
                                      float* __restrict__ invstd)
{
  int i = threadIdx.x;
  if (i >= BB * GG) return;
  const float N = 3.f * (float)DHW;             // 3 channels/group * D*H*W
  float m = sum[i] / N;
  float v = sq[i] / N - m * m;
  mean[i]   = m;
  invstd[i] = rsqrtf(v + 1e-5f);
}

// 4) fused GN + (optional residual) + SiLU, dense f32 -> padded bf16
__global__ __launch_bounds__(256) void gn_silu_apply_kernel(
    const float* __restrict__ Y, const float* __restrict__ mean,
    const float* __restrict__ invstd, const float* __restrict__ gamma,
    const float* __restrict__ beta, const bf16* __restrict__ resp,
    bf16* __restrict__ outp)
{
  size_t idx = (size_t)blockIdx.x * blockDim.x + threadIdx.x;
  if (idx >= (size_t)BB * AG * DHW) return;
  int bc = (int)(idx / DHW);                    // b*AG + c
  int c  = bc % AG;
  int b  = bc / AG;
  int r  = (int)(idx % DHW);
  int dz = r / HWXY;  int r2 = r % HWXY;
  int h  = r2 / WW;   int w  = r2 % WW;
  size_t pidx = (size_t)bc * PDHW + (size_t)(dz + 1) * PHW +
                (size_t)(h + 1) * PW + (w + 1);

  int g = c / 3;
  float x = (Y[idx] - mean[b * GG + g]) * invstd[b * GG + g];
  x = x * gamma[c] + beta[c];
  if (resp) x += (float)resp[pidx];
  outp[pidx] = (bf16)(x / (1.f + __expf(-x)));  // silu
}

// =====================================================================
// 5) head: 24->1 conv + softmax over D + expected offset + add d_coarse
//    (reads the padded tensor: no boundary branches at all)
// =====================================================================
__global__ __launch_bounds__(256) void head_softmax_kernel(
    const bf16* __restrict__ Xp, const float* __restrict__ wO,
    const float* __restrict__ bO, const float* __restrict__ dC,
    float* __restrict__ out)
{
  __shared__ float lw[AG * 27];
  __shared__ float lb;
  for (int i = threadIdx.x; i < AG * 27; i += blockDim.x) lw[i] = wO[i];
  if (threadIdx.x == 0) lb = bO[0];
  __syncthreads();

  int idx = blockIdx.x * blockDim.x + threadIdx.x;   // over B*H*W
  if (idx >= BB * HWXY) return;
  int w = idx % WW;  int t = idx / WW;
  int h = t % HH;    int b = t / HH;

  float logit[DDZ];
  for (int dd = 0; dd < DDZ; ++dd) {
    float acc = lb;
    for (int c = 0; c < AG; ++c) {
      const bf16* pc = Xp + (size_t)(b * AG + c) * PDHW +
                       (size_t)dd * PHW + (size_t)h * PW + w;  // tap (-1,-1,-1)
      const float* lwp = &lw[c * 27];
#pragma unroll
      for (int kd = 0; kd < 3; ++kd)
#pragma unroll
        for (int ky = 0; ky < 3; ++ky) {
          const bf16* prow = pc + kd * PHW + ky * PW;
#pragma unroll
          for (int kx = 0; kx < 3; ++kx)
            acc += (float)prow[kx] * lwp[kd * 9 + ky * 3 + kx];
        }
    }
    logit[dd] = acc;
  }
  float mx = logit[0];
#pragma unroll
  for (int dd = 1; dd < DDZ; ++dd) mx = fmaxf(mx, logit[dd]);
  float se = 0.f, so = 0.f;
#pragma unroll
  for (int dd = 0; dd < DDZ; ++dd) {
    float e = __expf(logit[dd] - mx);
    se += e;
    so += e * (float)(dd - HR);
  }
  out[idx] = dC[idx] + so / se;
}

// =====================================================================
// host orchestration
// =====================================================================
extern "C" void kernel_launch(void* const* d_in, const int* in_sizes, int n_in,
                              void* d_out, int out_size, void* d_ws, size_t ws_size,
                              hipStream_t stream)
{
  (void)in_sizes; (void)n_in; (void)out_size; (void)ws_size;

  const float* fL   = (const float*)d_in[0];
  const float* fR   = (const float*)d_in[1];
  const float* dC   = (const float*)d_in[2];
  const float* w_in = (const float*)d_in[3];
  const float* g_in = (const float*)d_in[4];
  const float* b_in = (const float*)d_in[5];
  const float* w_a1 = (const float*)d_in[6];
  const float* g_a1 = (const float*)d_in[7];
  const float* b_a1 = (const float*)d_in[8];
  const float* w_a2 = (const float*)d_in[9];
  const float* g_a2 = (const float*)d_in[10];
  const float* b_a2 = (const float*)d_in[11];
  const float* w_b1 = (const float*)d_in[12];
  const float* g_b1 = (const float*)d_in[13];
  const float* b_b1 = (const float*)d_in[14];
  const float* w_b2 = (const float*)d_in[15];
  const float* g_b2 = (const float*)d_in[16];
  const float* b_b2 = (const float*)d_in[17];
  const float* w_out = (const float*)d_in[18];
  const float* b_out = (const float*)d_in[19];
  float* outp = (float*)d_out;

  // ---- workspace carve-up (256-aligned)
  char* ws = (char*)d_ws;
  const size_t szCVP = ((size_t)BB * GG * PDHW * sizeof(bf16) + 255) & ~255ull;
  const size_t szXP  = ((size_t)BB * AG * PDHW * sizeof(bf16) + 255) & ~255ull;
  const size_t szY   = (size_t)BB * AG * DHW * sizeof(float);
  bf16*  CVp = (bf16*)(ws);
  bf16*  Xa  = (bf16*)(ws + szCVP);
  bf16*  Xb  = (bf16*)(ws + szCVP + szXP);
  bf16*  Xc  = (bf16*)(ws + szCVP + 2 * szXP);
  float* Yf  = (float*)(ws + szCVP + 3 * szXP);
  float* st  = (float*)(ws + szCVP + 3 * szXP + szY);
  float* s_sum = st, * s_sq = st + 32, * s_mean = st + 64, * s_inv = st + 96;

  // 0) zero the whole padded arena once per launch (pads stay zero; all
  //    interiors are overwritten before being read)
  {
    long n16 = (long)((szCVP + 3 * szXP) / 16);
    long grid = (n16 + 255) / 256;
    zero_arena_kernel<<<(int)grid, 256, 0, stream>>>((uint4*)ws, n16);
  }

  // 1) cost volume
  {
    int n = BB * DDZ * HWXY;                                       // 1,843,200
    warp_corr_kernel<<<(n + 255) / 256, 256, 0, stream>>>(fL, fR, dC, CVp);
  }

  const int conv_grid  = BB * 3600;                                // 14,400 blocks
  const int apply_n    = BB * AG * DHW;                            // 44,236,800
  const int apply_grid = (apply_n + 255) / 256;

  auto stage = [&](const bf16* Xin, const float* Wt, int CIN,
                   const float* gamma, const float* beta,
                   const bf16* res, bf16* Xout) {
    zero_stats_kernel<<<1, 64, 0, stream>>>(st);
    conv3d_wmma_kernel<<<conv_grid, 128, 0, stream>>>(Xin, Wt, Yf, s_sum, s_sq, CIN);
    finalize_stats_kernel<<<1, 64, 0, stream>>>(s_sum, s_sq, s_mean, s_inv);
    gn_silu_apply_kernel<<<apply_grid, 256, 0, stream>>>(Yf, s_mean, s_inv,
                                                         gamma, beta, res, Xout);
  };

  // 2) conv_in + GN + SiLU
  stage(CVp, w_in, GG, g_in, b_in, nullptr, Xa);
  // 3) res block A
  stage(Xa, w_a1, AG, g_a1, b_a1, nullptr, Xb);
  stage(Xb, w_a2, AG, g_a2, b_a2, Xa, Xc);          // silu(x + gn(conv))
  // 4) res block B
  stage(Xc, w_b1, AG, g_b1, b_b1, nullptr, Xb);
  stage(Xb, w_b2, AG, g_b2, b_b2, Xc, Xa);
  // 5) head conv + softmax + expected offset
  {
    int n = BB * HWXY;                                             // 204,800
    head_softmax_kernel<<<(n + 255) / 256, 256, 0, stream>>>(Xa, w_out, b_out,
                                                             dC, outp);
  }
}
